// ConcatLSTMLayer_57062935494749
// MI455X (gfx1250) — matrix operand order repaired and checked by reading, not compiled, not run on hardware
//
#include <hip/hip_runtime.h>
#include <hip/hip_bf16.h>
#include <math.h>

// ---------------------------------------------------------------------------
// ConcatLSTM for MI455X (gfx1250, wave32, WMMA bf16 w/ f32 accumulation)
//   T=512, B=64, I=512, H=512, 4H=2048, C=128
// LDS-staged weights + software-pipelined fragment loads:
// hot loops are ds_load_b128 (next iter) -> v_wmma_f32_16x16x32_bf16 (cur iter)
// ---------------------------------------------------------------------------

#define T_STEPS 512
#define BATCH   64
#define IN_DIM  512
#define HID     512
#define GATES   2048   // 4*HID
#define CTX     128
#define KPAD    (HID + 8)   // 520 bf16 per LDS row = 1040 B (16B aligned, bank-conflict free)

typedef __attribute__((ext_vector_type(16))) __bf16 v16bf;
typedef __attribute__((ext_vector_type(8)))  __bf16 v8bf;
typedef __attribute__((ext_vector_type(8)))  float  v8f;
typedef __attribute__((ext_vector_type(4)))  float  v4f;

// ---- optional CDNA5 async global->LDS staging (ASYNCcnt path) -------------
#if defined(__has_builtin)
#  if __has_builtin(__builtin_amdgcn_global_load_async_to_lds_b128) && \
      __has_builtin(__builtin_amdgcn_s_wait_asynccnt)
#    define USE_ASYNC_LDS 1
#  endif
#endif
#ifndef USE_ASYNC_LDS
#  define USE_ASYNC_LDS 0
#endif

// The builtin's params are pointers to int4 (vector_size(16)) in AS(1)/AS(3):
// clang prints AS(1) as "__device__" and AS(3) as "__shared__" in HIP mode.
typedef int v4i_vs __attribute__((vector_size(16)));
typedef __attribute__((address_space(1))) v4i_vs gas_v4i;
typedef __attribute__((address_space(3))) v4i_vs las_v4i;

// copy 16 bytes global -> LDS (async DMA if available, else load+ds_store)
__device__ __forceinline__ void copy16_g2l(__bf16* lptr, const __bf16* gptr) {
#if USE_ASYNC_LDS
    // int->pointer casts; LDS flat addr -> AS(3) uses addr[31:0] (aperture rule).
    __builtin_amdgcn_global_load_async_to_lds_b128(
        (gas_v4i*)(unsigned long long)(uintptr_t)gptr,
        (las_v4i*)(unsigned int)(uintptr_t)lptr,
        0, 0);
#else
    *(v8bf*)lptr = *(const v8bf*)gptr;
#endif
}

__device__ __forceinline__ void stage_fence() {
#if USE_ASYNC_LDS
    __builtin_amdgcn_s_wait_asynccnt(0);
#endif
    __syncthreads();
}

__device__ __forceinline__ float sigmoidf_(float x) {
    return 1.0f / (1.0f + __expf(-x));
}

// A fragment (16x32 bf16, MxK) from fp32 row-major source, converted on the
// fly (v_cvt_pk_bf16_f32 co-executes with XDL WMMA). ISA layout: lane
// (lh = lane/16, ln = lane%16) holds row M = ln,
// K = {8*lh..8*lh+7} U {16+8*lh..16+8*lh+7}   (two contiguous 32B chunks).
__device__ __forceinline__ v16bf load_a_frag_f32(const float* __restrict__ rowPtr,
                                                 int kb, int lh) {
    const v4f* p0 = (const v4f*)(rowPtr + kb + 8 * lh);
    const v4f* p1 = (const v4f*)(rowPtr + kb + 16 + 8 * lh);
    v4f c0 = p0[0], c1 = p0[1], c2 = p1[0], c3 = p1[1];
    v16bf a;
#pragma unroll
    for (int e = 0; e < 4; ++e) {
        a[e]      = (__bf16)c0[e];
        a[4 + e]  = (__bf16)c1[e];
        a[8 + e]  = (__bf16)c2[e];
        a[12 + e] = (__bf16)c3[e];
    }
    return a;
}

// Same A fragment from a bf16 row-major source (recurrent h matrix).
__device__ __forceinline__ v16bf load_a_frag_bf16(const __bf16* __restrict__ rowPtr,
                                                  int kb, int lh) {
    v8bf lo = *(const v8bf*)(rowPtr + kb + 8 * lh);        // global_load_b128
    v8bf hi = *(const v8bf*)(rowPtr + kb + 16 + 8 * lh);
    v16bf a;
#pragma unroll
    for (int e = 0; e < 8; ++e) { a[e] = lo[e]; a[8 + e] = hi[e]; }
    return a;
}

// B fragment (32x16 bf16, KxN) read from an LDS-staged weight tile.
// Weight row-major [rows, K] (GEMM is X @ W^T, so B(k,n) = W[n][k]).
// ISA layout: lane holds column n = ln, K = {8*lh..8*lh+7} U {16+8*lh..16+8*lh+7}
//   -> two 16B chunks of LDS row `row`  (two ds_load_b128).
__device__ __forceinline__ v16bf load_b_frag_lds(const __bf16* lds, int row,
                                                 int kb, int lh) {
    const __bf16* p = lds + (size_t)row * KPAD + kb + 8 * lh;
    v8bf lo = *(const v8bf*)p;
    v8bf hi = *(const v8bf*)(p + 16);
    v16bf b;
#pragma unroll
    for (int e = 0; e < 8; ++e) { b[e] = lo[e]; b[8 + e] = hi[e]; }
    return b;
}

__device__ __forceinline__ v8f wmma_bf16(v16bf a, v16bf b, v8f c) {
    // 8 args: (neg_a, A, neg_b, B, c_mod, C, reuse_a, reuse_b)
    return __builtin_amdgcn_wmma_f32_16x16x32_bf16(false, a, false, b,
                                                   (short)0, c, false, false);
}

// ---------------------------------------------------------------------------
// Prep kernels
// ---------------------------------------------------------------------------
__global__ void cvt_bf16_kernel(const float* __restrict__ src,
                                __bf16* __restrict__ dst, int n) {
    int i = blockIdx.x * blockDim.x + threadIdx.x;
    if (i < n) dst[i] = (__bf16)src[i];
}

// bias_all[b][g] = date_contexts[b,:] . weight_mh[g,:] + bias_ih[g] + bias_hh[g]
__global__ void bias_md_kernel(const float* __restrict__ dc,
                               const float* __restrict__ wmh,
                               const float* __restrict__ bih,
                               const float* __restrict__ bhh,
                               float* __restrict__ bias_all) {
    int idx = blockIdx.x * blockDim.x + threadIdx.x;
    if (idx >= BATCH * GATES) return;
    int b = idx / GATES, g = idx % GATES;
    const float* dcr = dc + (size_t)b * CTX;
    const float* wr  = wmh + (size_t)g * CTX;
    float s = 0.f;
#pragma unroll 8
    for (int c = 0; c < CTX; ++c) s = fmaf(dcr[c], wr[c], s);
    bias_all[idx] = s + bih[g] + bhh[g];
}

__global__ void init_state_kernel(const float* __restrict__ h0,
                                  const float* __restrict__ c0,
                                  __bf16* __restrict__ hbuf,  // buffer 0 of 2
                                  float* __restrict__ cbuf,
                                  unsigned* __restrict__ barrier) {
    int i = blockIdx.x * blockDim.x + threadIdx.x;
    if (i == 0) *barrier = 0u;                 // deterministic per launch
    if (i < BATCH * HID) {
        hbuf[i] = (__bf16)h0[i];
        cbuf[i] = c0[i];
    }
}

// ---------------------------------------------------------------------------
// Phase 1: x_gates GEMM.  M = T*B = 32768, N = 4H = 2048, K = I = 512.
// Block = 8 waves sharing one 64-column group of W_ih (staged in 65KB LDS),
// each wave owns a 16x64 output strip. A+B fragments are double-buffered so
// iteration k+1's loads overlap iteration k's 4 WMMAs.
// grid: 256 m-groups x 32 n-groups = 8192 blocks.
// ---------------------------------------------------------------------------
__global__ __launch_bounds__(256)
void xgates_kernel(const float* __restrict__ x,        // [T*B, I] fp32
                   const __bf16* __restrict__ Wih,     // [4H, I] bf16
                   const float* __restrict__ bias_all, // [B, 4H] fp32
                   float* __restrict__ xg) {           // [T*B, 4H] fp32
    extern __shared__ __align__(16) char smem_raw[];
    __bf16* sB = (__bf16*)smem_raw;           // [64][KPAD]

    const int lane = threadIdx.x & 31;
    const int wave = threadIdx.x >> 5;
    const int ng = blockIdx.x & 31;           // 0..31  column group (64 cols)
    const int mg = blockIdx.x >> 5;           // 0..255 row group (8 tiles)
    const int n0 = ng * 64;
    const int m0 = (mg * 8 + wave) * 16;
    const int lh = lane >> 4;
    const int ln = lane & 15;

    // ---- stage 64 rows x 512 K of W_ih into LDS (16B chunks, coalesced) ----
    for (int c = threadIdx.x; c < 64 * (IN_DIM / 8); c += 256) {
        const int r  = c >> 6;                 // 64 chunks per row
        const int kc = (c & 63) * 8;
        copy16_g2l(&sB[(size_t)r * KPAD + kc],
                   &Wih[(size_t)(n0 + r) * IN_DIM + kc]);
    }
    stage_fence();

    const float* rowPtr = x + (size_t)(m0 + ln) * IN_DIM;

    // ---- software-pipelined K loop ----
    v16bf aC  = load_a_frag_f32(rowPtr, 0, lh);
    v16bf bC0 = load_b_frag_lds(sB,  0 + ln, 0, lh);
    v16bf bC1 = load_b_frag_lds(sB, 16 + ln, 0, lh);
    v16bf bC2 = load_b_frag_lds(sB, 32 + ln, 0, lh);
    v16bf bC3 = load_b_frag_lds(sB, 48 + ln, 0, lh);
    v8f acc0 = {}, acc1 = {}, acc2 = {}, acc3 = {};
#pragma unroll
    for (int kb = 0; kb < IN_DIM; kb += 32) {
        const int kn = kb + 32;
        v16bf aN = aC, bN0 = bC0, bN1 = bC1, bN2 = bC2, bN3 = bC3;
        if (kn < IN_DIM) {
            __builtin_prefetch(rowPtr + kn + 32, 0, 3);   // global_prefetch_b8
            aN  = load_a_frag_f32(rowPtr, kn, lh);
            bN0 = load_b_frag_lds(sB,  0 + ln, kn, lh);
            bN1 = load_b_frag_lds(sB, 16 + ln, kn, lh);
            bN2 = load_b_frag_lds(sB, 32 + ln, kn, lh);
            bN3 = load_b_frag_lds(sB, 48 + ln, kn, lh);
        }
        acc0 = wmma_bf16(aC, bC0, acc0);
        acc1 = wmma_bf16(aC, bC1, acc1);
        acc2 = wmma_bf16(aC, bC2, acc2);
        acc3 = wmma_bf16(aC, bC3, acc3);
        aC = aN; bC0 = bN0; bC1 = bN1; bC2 = bN2; bC3 = bN3;
    }

    v8f accs[4] = {acc0, acc1, acc2, acc3};
#pragma unroll
    for (int j = 0; j < 4; ++j) {
        const int n = n0 + j * 16 + ln;
#pragma unroll
        for (int r = 0; r < 8; ++r) {
            const int m = m0 + r + 8 * lh;        // global row in [0, T*B)
            const int b = m & (BATCH - 1);        // row -> batch index
            xg[(size_t)m * GATES + n] = accs[j][r] + bias_all[(size_t)b * GATES + n];
        }
    }
}

// ---------------------------------------------------------------------------
// Phase 2: persistent recurrent scan. 16 blocks x 8 waves = 128 waves = all
// 4(batch) x 32(hidden) state tiles. Block covers 2 hidden-column tiles x 4
// batch tiles; the 128 W_hh rows it needs are staged ONCE in 130KB LDS before
// the 512-step loop. Per step: 16 pipelined K-iterations x 4 WMMAs fed by
// ds_load_b128, cell update, double-buffered bf16 h, device-scope barrier.
// ---------------------------------------------------------------------------
__global__ __launch_bounds__(256)
void lstm_scan_kernel(const float* __restrict__ xg,    // [T, B, 4H] fp32
                      const __bf16* __restrict__ Whh,  // [4H, H] bf16
                      __bf16* __restrict__ hbuf,       // [2][B, H] bf16 (ping-pong)
                      float* __restrict__ cbuf,        // [B, H] fp32
                      float* __restrict__ out,         // outputs | hT | cT
                      unsigned* __restrict__ barrier) {
    extern __shared__ __align__(16) char smem_raw[];
    __bf16* sW = (__bf16*)smem_raw;           // [128][KPAD]

    const int lane = threadIdx.x & 31;
    const int wave = threadIdx.x >> 5;
    const int ntl  = wave >> 2;               // 0..1 local hidden tile
    const int mt   = wave & 3;                // 0..3 batch tile
    const int nt   = blockIdx.x * 2 + ntl;    // 0..31 global hidden tile
    const int b0 = mt * 16;
    const int n0 = nt * 16;
    const int lh = lane >> 4;
    const int ln = lane & 15;
    const int nblocks = (int)gridDim.x;

    // ---- stage the block's 128 W_hh rows into LDS (loop-invariant!) ----
    // local row rl = ntl*64 + j*16 + c  <->  global row j*512 + nt*16 + c
    for (int c = threadIdx.x; c < 128 * (HID / 8); c += 256) {
        const int rl  = c >> 6;                // 64 chunks per row
        const int kc  = (c & 63) * 8;
        const int tl  = rl >> 6;               // 0..1
        const int j   = (rl >> 4) & 3;         // gate
        const int col = rl & 15;
        const int grow = j * HID + (blockIdx.x * 2 + tl) * 16 + col;
        copy16_g2l(&sW[(size_t)rl * KPAD + kc],
                   &Whh[(size_t)grow * HID + kc]);
    }
    stage_fence();

    const int bRow  = b0 + ln;                 // A-fragment row (batch)
    const int nCol  = n0 + ln;                 // C/D column (hidden unit)
    const int rowB0 = ntl * 64 + ln;           // LDS row for gate 0 fragment

    for (int t = 0; t < T_STEPS; ++t) {
        const __bf16* hsrc = hbuf + (size_t)(t & 1) * (BATCH * HID);
        __bf16*       hdst = hbuf + (size_t)((t + 1) & 1) * (BATCH * HID);
        const __bf16* hrow = hsrc + (size_t)bRow * HID;

        const float* xgt = xg + (size_t)t * BATCH * GATES;
        // warm the xg lines this wave will read in the epilogue
#pragma unroll
        for (int j = 0; j < 4; ++j)
            __builtin_prefetch(xgt + (size_t)bRow * GATES + j * HID + n0, 0, 3);

        // ---- software-pipelined K loop ----
        v16bf aC  = load_a_frag_bf16(hrow, 0, lh);
        v16bf bC0 = load_b_frag_lds(sW, rowB0 +  0, 0, lh);
        v16bf bC1 = load_b_frag_lds(sW, rowB0 + 16, 0, lh);
        v16bf bC2 = load_b_frag_lds(sW, rowB0 + 32, 0, lh);
        v16bf bC3 = load_b_frag_lds(sW, rowB0 + 48, 0, lh);
        v8f acc0 = {}, acc1 = {}, acc2 = {}, acc3 = {};
#pragma unroll
        for (int kb = 0; kb < HID; kb += 32) {
            const int kn = kb + 32;
            v16bf aN = aC, bN0 = bC0, bN1 = bC1, bN2 = bC2, bN3 = bC3;
            if (kn < HID) {
                aN  = load_a_frag_bf16(hrow, kn, lh);
                bN0 = load_b_frag_lds(sW, rowB0 +  0, kn, lh);
                bN1 = load_b_frag_lds(sW, rowB0 + 16, kn, lh);
                bN2 = load_b_frag_lds(sW, rowB0 + 32, kn, lh);
                bN3 = load_b_frag_lds(sW, rowB0 + 48, kn, lh);
            }
            acc0 = wmma_bf16(aC, bC0, acc0);
            acc1 = wmma_bf16(aC, bC1, acc1);
            acc2 = wmma_bf16(aC, bC2, acc2);
            acc3 = wmma_bf16(aC, bC3, acc3);
            aC = aN; bC0 = bN0; bC1 = bN1; bC2 = bN2; bC3 = bN3;
        }

        const v8f accs[4] = {acc0, acc1, acc2, acc3};
#pragma unroll
        for (int r = 0; r < 8; ++r) {
            const int b = b0 + r + 8 * lh;
            const float* xgb = xgt + (size_t)b * GATES;
            const float gi = accs[0][r] + xgb[0 * HID + nCol];
            const float gf = accs[1][r] + xgb[1 * HID + nCol];
            const float gg = accs[2][r] + xgb[2 * HID + nCol];
            const float go = accs[3][r] + xgb[3 * HID + nCol];
            const float i_ = sigmoidf_(gi);
            const float f_ = sigmoidf_(gf);
            const float g_ = tanhf(gg);
            const float o_ = sigmoidf_(go);
            const size_t cidx = (size_t)b * HID + nCol;
            const float c_ = f_ * cbuf[cidx] + i_ * g_;
            cbuf[cidx] = c_;
            const float h_ = o_ * tanhf(c_);
            out[(size_t)t * BATCH * HID + cidx] = h_;   // outputs[t]
            hdst[cidx] = (__bf16)h_;
            if (t == T_STEPS - 1) {
                out[(size_t)T_STEPS * BATCH * HID + cidx] = h_;                 // hT
                out[(size_t)T_STEPS * BATCH * HID + BATCH * HID + cidx] = c_;   // cT
            }
        }

        // ---- device-scope barrier between timesteps (monotonic counter) ----
        __threadfence();            // release: make h/c writes visible (agent)
        __syncthreads();
        if (threadIdx.x == 0) {
            atomicAdd(barrier, 1u);
            const unsigned target = (unsigned)nblocks * (unsigned)(t + 1);
            volatile unsigned* vb = (volatile unsigned*)barrier;
            while (*vb < target) __builtin_amdgcn_s_sleep(1);
        }
        __syncthreads();
        __threadfence();            // acquire: invalidate stale WGP$ lines
    }
}

// ---------------------------------------------------------------------------
// Launch
// ---------------------------------------------------------------------------
extern "C" void kernel_launch(void* const* d_in, const int* in_sizes, int n_in,
                              void* d_out, int out_size, void* d_ws, size_t ws_size,
                              hipStream_t stream) {
    (void)in_sizes; (void)n_in; (void)out_size; (void)ws_size;
    const float* x   = (const float*)d_in[0];  // [T, B, I]
    const float* dc  = (const float*)d_in[1];  // [B, C]
    const float* h0  = (const float*)d_in[2];  // [B, H]
    const float* c0  = (const float*)d_in[3];  // [B, H]
    const float* Wih = (const float*)d_in[4];  // [4H, I]
    const float* Whh = (const float*)d_in[5];  // [4H, H]
    const float* Wmh = (const float*)d_in[6];  // [4H, C]
    const float* bih = (const float*)d_in[7];  // [4H]
    const float* bhh = (const float*)d_in[8];  // [4H]
    float* out = (float*)d_out;

    char* ws = (char*)d_ws;
    size_t off = 0;
    unsigned* barrier  = (unsigned*)(ws + off); off += 256;
    float*    bias_all = (float*)(ws + off);    off += (size_t)BATCH * GATES * 4;
    __bf16*   WihB     = (__bf16*)(ws + off);   off += (size_t)GATES * IN_DIM * 2;
    __bf16*   WhhB     = (__bf16*)(ws + off);   off += (size_t)GATES * HID * 2;
    __bf16*   hbuf     = (__bf16*)(ws + off);   off += (size_t)2 * BATCH * HID * 2;
    float*    cbuf     = (float*)(ws + off);    off += (size_t)BATCH * HID * 4;
    float*    xg       = (float*)(ws + off);    off += (size_t)T_STEPS * BATCH * GATES * 4;
    // total workspace ~273 MB

    cvt_bf16_kernel<<<(GATES * IN_DIM + 255) / 256, 256, 0, stream>>>(Wih, WihB, GATES * IN_DIM);
    cvt_bf16_kernel<<<(GATES * HID + 255) / 256, 256, 0, stream>>>(Whh, WhhB, GATES * HID);
    bias_md_kernel<<<(BATCH * GATES + 255) / 256, 256, 0, stream>>>(dc, Wmh, bih, bhh, bias_all);
    init_state_kernel<<<(BATCH * HID + 255) / 256, 256, 0, stream>>>(h0, c0, hbuf, cbuf, barrier);

    // Phase 1: 8192 blocks, 8 waves each; 65KB dynamic LDS for the W_ih tile.
    const size_t lds_xg = (size_t)64 * KPAD * sizeof(__bf16);     // 66,560 B
    xgates_kernel<<<8192, 256, lds_xg, stream>>>(x, WihB, bias_all, xg);

    // Phase 2: persistent scan, 16 blocks; 130KB dynamic LDS for W_hh rows.
    const size_t lds_sc = (size_t)128 * KPAD * sizeof(__bf16);    // 133,120 B
    lstm_scan_kernel<<<16, 256, lds_sc, stream>>>(xg, WhhB, hbuf, cbuf, out, barrier);
}
